// MassivePool_61546881351933
// MI455X (gfx1250) — compile-verified, matrix operand order checked by reading, not run on hardware
//
#include <hip/hip_runtime.h>
#include <hip/hip_bf16.h>
#include <stdint.h>

// ---------------- problem constants ----------------
constexpr int NQ = 1024;          // 2*512 queries
constexpr int D  = 1024;          // hidden dim
constexpr int P  = 200000;        // pool size
constexpr int K  = 128;           // top-k
constexpr int QTILE  = 128;       // queries per workgroup
constexpr int PTILE  = 256;       // pool rows per workgroup
constexpr int PTILES = (P + PTILE - 1) / PTILE;          // 782
constexpr long long SSTRIDE = (long long)PTILES * PTILE; // 200192 (padded row)
constexpr int KSTEPS = D / 32;                           // 32

// ---------------- WMMA types ----------------
typedef __attribute__((ext_vector_type(16))) __bf16 v16bf;
typedef __attribute__((ext_vector_type(8)))  float  v8f;

union FragU { v16bf v; uint32_t u[8]; };

// pack two fp32 (as raw bits) into one dword of two bf16 (truncation): 1 x v_perm_b32
__device__ __forceinline__ uint32_t pk_bf16(uint32_t lo, uint32_t hi) {
  return __builtin_amdgcn_perm(hi, lo, 0x07060302u);   // bytes: lo[2],lo[3],hi[2],hi[3]
}

// fp32 -> order-preserving u16 key (bf16 truncation, sign-flipped: larger score => larger key)
__device__ __forceinline__ unsigned short f32_key(float f) {
  uint32_t u = __float_as_uint(f);
  unsigned short b = (unsigned short)(u >> 16);
  return (b & 0x8000u) ? (unsigned short)(~b) : (unsigned short)(b | 0x8000u);
}

// Build one 16x32 bf16 fragment from 32 contiguous fp32 (base already includes +hi*8 floats).
// VGPR layout (ISA 7.12.2, 16-bit A/B): lanes<16: dwords0-3 = K0..7, dwords4-7 = K16..23;
// lanes>=16: dwords0-3 = K8..15, dwords4-7 = K24..31 => two contiguous 8-float groups.
__device__ __forceinline__ v16bf make_frag(const uint32_t* base) {
  const uint4* p = (const uint4*)base;
  uint4 a0 = p[0];   // +0 floats
  uint4 a1 = p[1];   // +4
  uint4 a2 = p[4];   // +16
  uint4 a3 = p[5];   // +20
  FragU f;
  f.u[0] = pk_bf16(a0.x, a0.y); f.u[1] = pk_bf16(a0.z, a0.w);
  f.u[2] = pk_bf16(a1.x, a1.y); f.u[3] = pk_bf16(a1.z, a1.w);
  f.u[4] = pk_bf16(a2.x, a2.y); f.u[5] = pk_bf16(a2.z, a2.w);
  f.u[6] = pk_bf16(a3.x, a3.y); f.u[7] = pk_bf16(a3.z, a3.w);
  return f.v;
}

// ---------------- CDNA5 async global->LDS (guarded) ----------------
#if __has_builtin(__builtin_amdgcn_global_load_async_to_lds_b128) && \
    __has_builtin(__builtin_amdgcn_s_wait_asynccnt)
#define USE_ASYNC_LDS 1
#else
#define USE_ASYNC_LDS 0
#endif

typedef int v4i __attribute__((vector_size(16)));
typedef __attribute__((address_space(1))) v4i* g_v4i_ptr;   // global (AS1)
typedef __attribute__((address_space(3))) v4i* l_v4i_ptr;   // LDS (AS3)

// ---------------- kernel 1: scores GEMM (bf16 WMMA, async-LDS staged B) ----------------
// grid (8 query tiles, 782 pool tiles), block 256 = 8 waves.
// wave w: 32-query M-group (w&3) of the 128-query tile, pool half (w>>2) of the 256-row tile.
// Per k-step each B fragment feeds TWO WMMAs (two A fragments) -> 16 WMMA / wave / k-step.
__global__ __launch_bounds__(256)
void gemm_scores(const float* __restrict__ Q, const float* __restrict__ E,
                 unsigned short* __restrict__ S) {
  __shared__ uint32_t Bbuf[2][PTILE * 32];   // 2 x 32KB fp32 tiles (double buffered)

  const int qt   = blockIdx.x;
  const int pt   = blockIdx.y;
  const int tid  = threadIdx.x;
  const int w    = tid >> 5;
  const int lane = tid & 31;
  const int hi   = lane >> 4;
  const int l16  = lane & 15;

  const int qbase   = qt * QTILE + (w & 3) * 32;   // 32 queries per wave
  const int rowbase = (w >> 2) * 128;              // wave's half of the 256-row tile

  v8f acc[16];                                     // [mf*8 + nf], mf in {0,1}
  const v8f vz = {0.f, 0.f, 0.f, 0.f, 0.f, 0.f, 0.f, 0.f};
#pragma unroll
  for (int i = 0; i < 16; ++i) acc[i] = vz;

  // per-thread staging: thread copies one pool row's 32 fp32 (128B) per k-step
  int pr_mine = pt * PTILE + tid;
  if (pr_mine >= P) pr_mine = P - 1;               // clamp (ragged last pool tile)
  const uint32_t* gsrc_row = (const uint32_t*)E + (size_t)pr_mine * D;
  uint32_t* lds_row_0 = &Bbuf[0][tid * 32];
  uint32_t* lds_row_1 = &Bbuf[1][tid * 32];

  auto issue_tile = [&](int kt) {
    const uint32_t* g = gsrc_row + kt * 32;
    uint32_t* l = (kt & 1) ? lds_row_1 : lds_row_0;
#if USE_ASYNC_LDS
#pragma unroll
    for (int j = 0; j < 8; ++j) {
      __builtin_amdgcn_global_load_async_to_lds_b128(
          (g_v4i_ptr)(g + j * 4), (l_v4i_ptr)(l + j * 4), 0, 0);
    }
#else
#pragma unroll
    for (int j = 0; j < 8; ++j) {
      uint4 v = ((const uint4*)g)[j];
      ((uint4*)l)[j] = v;
    }
#endif
  };

  const uint32_t* qptr0 = (const uint32_t*)Q + (size_t)(qbase + l16) * D + hi * 8;
  const uint32_t* qptr1 = (const uint32_t*)Q + (size_t)(qbase + 16 + l16) * D + hi * 8;

  issue_tile(0);
#pragma unroll 1
  for (int kt = 0; kt < KSTEPS; ++kt) {
#if USE_ASYNC_LDS
    if (kt + 1 < KSTEPS) {
      issue_tile(kt + 1);
      __builtin_amdgcn_s_wait_asynccnt(8);   // tile kt's 8 copies complete (in-order)
    } else {
      __builtin_amdgcn_s_wait_asynccnt(0);
    }
#else
    if (kt + 1 < KSTEPS) issue_tile(kt + 1);
#endif
    __syncthreads();                                   // all waves' portions visible

    v16bf a0 = make_frag(qptr0 + kt * 32);             // queries: direct global (L2-hot)
    v16bf a1 = make_frag(qptr1 + kt * 32);

    const uint32_t* bbase = (kt & 1) ? &Bbuf[1][0] : &Bbuf[0][0];
#pragma unroll
    for (int nf = 0; nf < 8; ++nf) {
      const uint32_t* lrow = bbase + (size_t)(rowbase + nf * 16 + l16) * 32 + hi * 8;
      v16bf b = make_frag(lrow);                       // ds_load_b128 x4 + 8 perms
      acc[nf] = __builtin_amdgcn_wmma_f32_16x16x32_bf16(
          false, a0, false, b, (short)0, acc[nf], false, false);
      acc[8 + nf] = __builtin_amdgcn_wmma_f32_16x16x32_bf16(
          false, a1, false, b, (short)0, acc[8 + nf], false, false);
    }
    __syncthreads();                                   // before buffer is overwritten
  }

  // ---- transpose keys through LDS (reuse Bbuf as staging: 8 waves x 32x128 u16 = 64KB)
  unsigned short(*stage)[32][128] = (unsigned short(*)[32][128]) & Bbuf[0][0];
  const int pbase = pt * PTILE + rowbase;
  // C layout: element e of v8f, lanes<16 -> M=e, lanes>=16 -> M=e+8; N = lane&15.
#pragma unroll
  for (int mf = 0; mf < 2; ++mf)
#pragma unroll
    for (int nf = 0; nf < 8; ++nf)
#pragma unroll
      for (int e = 0; e < 8; ++e)
        stage[w][mf * 16 + e + 8 * hi][nf * 16 + l16] = f32_key(acc[mf * 8 + nf][e]);

  __syncthreads();

  // coalesced store: 32 lanes x 8B cover one 128-key row
  for (int r = 0; r < 32; ++r) {
    uint2 d = *(const uint2*)&stage[w][r][lane * 4];
    *(uint2*)(S + (size_t)(qbase + r) * SSTRIDE + pbase + lane * 4) = d;
  }
}

// ---------------- kernel 2: per-query top-128 selection ----------------
// One 256-thread WG per query. 12-bit coarse histogram -> threshold bin,
// then collect+rank (key desc, index asc — matches jax top_k tie order).
constexpr int HBINS = 4096;
constexpr int EQCAP = 1024;

__global__ __launch_bounds__(256)
void select_topk(const unsigned short* __restrict__ S, int* __restrict__ TI) {
  __shared__ unsigned int hist[HBINS];           // 16 KB
  __shared__ unsigned int chunkSum[256];
  __shared__ unsigned int s_thr, s_cgt;
  __shared__ unsigned short gkey[128];
  __shared__ int            gidx[128];
  __shared__ unsigned short ekey[EQCAP];
  __shared__ int            eidx[EQCAP];
  __shared__ unsigned int   gcnt, ecnt;

  const int q = blockIdx.x;
  const int t = threadIdx.x;
  const unsigned short* row = S + (size_t)q * SSTRIDE;

  for (int i = t; i < HBINS; i += 256) hist[i] = 0;
  if (t == 0) { gcnt = 0; ecnt = 0; }
  __syncthreads();

  // histogram over top 12 key bits, vectorized 8 keys / uint4
  const uint4* vrow = (const uint4*)row;         // rows are 16B aligned (SSTRIDE*2 % 16 == 0)
  for (int i = t; i < P / 8; i += 256) {
    uint4 v = vrow[i];
    atomicAdd(&hist[(v.x & 0xFFFFu) >> 4], 1u); atomicAdd(&hist[(v.x >> 16) >> 4], 1u);
    atomicAdd(&hist[(v.y & 0xFFFFu) >> 4], 1u); atomicAdd(&hist[(v.y >> 16) >> 4], 1u);
    atomicAdd(&hist[(v.z & 0xFFFFu) >> 4], 1u); atomicAdd(&hist[(v.z >> 16) >> 4], 1u);
    atomicAdd(&hist[(v.w & 0xFFFFu) >> 4], 1u); atomicAdd(&hist[(v.w >> 16) >> 4], 1u);
  }
  __syncthreads();

  { unsigned s = 0;
    for (int j = 0; j < 16; ++j) s += hist[t * 16 + j];
    chunkSum[t] = s; }
  __syncthreads();

  if (t == 0) {
    unsigned accu = 0; int c = 255;
    for (; c > 0; --c) { if (accu + chunkSum[c] >= (unsigned)K) break; accu += chunkSum[c]; }
    int b = c * 16 + 15;
    for (; b > c * 16; --b) { if (accu + hist[b] >= (unsigned)K) break; accu += hist[b]; }
    s_thr = (unsigned)b;      // threshold bin
    s_cgt = accu;             // #keys strictly above bin b  (<= 127)
  }
  __syncthreads();
  const unsigned thr = s_thr;
  const unsigned cgt = s_cgt;

  for (int i = t; i < K; i += 256) TI[q * K + i] = 0;   // defined values in all slots

  for (int i = t; i < P; i += 256) {
    unsigned k = row[i];
    unsigned bin = k >> 4;
    if (bin > thr) {
      unsigned pos = atomicAdd(&gcnt, 1u);
      if (pos < 128u) { gkey[pos] = (unsigned short)k; gidx[pos] = i; }
    } else if (bin == thr) {
      unsigned pos = atomicAdd(&ecnt, 1u);
      if (pos < (unsigned)EQCAP) { ekey[pos] = (unsigned short)k; eidx[pos] = i; }
    }
  }
  __syncthreads();

  const int ng = (int)(cgt < 127u ? cgt : 127u);
  if (t < ng) {
    unsigned short mk = gkey[t]; int mi = gidx[t]; int r = 0;
    for (int j = 0; j < ng; ++j) {
      unsigned short k2 = gkey[j]; int i2 = gidx[j];
      r += (k2 > mk) || (k2 == mk && i2 < mi);
    }
    TI[q * K + r] = mi;
  }
  const int need = K - ng;
  const int m = (int)(ecnt < (unsigned)EQCAP ? ecnt : (unsigned)EQCAP);
  for (int x = t; x < m; x += 256) {
    unsigned short mk = ekey[x]; int mi = eidx[x]; int r = 0;
    for (int j = 0; j < m; ++j) {
      unsigned short k2 = ekey[j]; int i2 = eidx[j];
      r += (k2 > mk) || (k2 == mk && i2 < mi);
    }
    if (r < need) TI[q * K + ng + r] = mi;
  }
}

// ---------------- kernel 3: masked gather ----------------
// one 256-thread WG per (query, slot); 1024 floats = 256 x float4
__global__ __launch_bounds__(256)
void gather_out(const float* __restrict__ E, const int* __restrict__ TI,
                const int* __restrict__ KP, float* __restrict__ O) {
  const int b = blockIdx.x;
  const int q = b >> 7;
  const int slot = b & 127;
  const int t = threadIdx.x;
  float4* orow = (float4*)(O + (size_t)b * D);
  if (slot < KP[q]) {
    const int idx = TI[q * K + slot];
    const float4* e = (const float4*)(E + (size_t)idx * D);
    orow[t] = e[t];
  } else {
    orow[t] = make_float4(0.f, 0.f, 0.f, 0.f);
  }
}

// ---------------- launcher ----------------
// inputs: [0] query_hidden f32 [2,512,1024], [1] embeddings f32 [200000,1024],
//         [2] k_predicted i32 [2,512], [3] phase_idx i32 (unused by reference)
// ws layout: [ u16 keys NQ*SSTRIDE (≈410 MB) | int top_idx NQ*K ]
extern "C" void kernel_launch(void* const* d_in, const int* in_sizes, int n_in,
                              void* d_out, int out_size, void* d_ws, size_t ws_size,
                              hipStream_t stream) {
  const float* Q  = (const float*)d_in[0];
  const float* E  = (const float*)d_in[1];
  const int*   KP = (const int*)d_in[2];
  (void)in_sizes; (void)n_in; (void)out_size; (void)ws_size;

  unsigned short* S  = (unsigned short*)d_ws;
  int*            TI = (int*)((char*)d_ws + (size_t)NQ * SSTRIDE * sizeof(unsigned short));
  float*          O  = (float*)d_out;

  gemm_scores<<<dim3(NQ / QTILE, PTILES), 256, 0, stream>>>(Q, E, S);
  select_topk<<<NQ, 256, 0, stream>>>(S, TI);
  gather_out<<<NQ * K, 256, 0, stream>>>(E, TI, KP, O);
}
